// custom_rnn_layer_901943132821
// MI455X (gfx1250) — matrix operand order repaired and checked by reading, not compiled, not run on hardware
//
#include <hip/hip_runtime.h>
#include <hip/hip_bf16.h>
#include <math.h>

// CDNA5 wave32 WMMA types
typedef __attribute__((ext_vector_type(16))) _Float16 v16h;
typedef __attribute__((ext_vector_type(8)))  _Float16 v8h;
typedef __attribute__((ext_vector_type(8)))  float    v8f;

// ---------------------------------------------------------------------------
// Fragment loaders (16-bit, wave32; contiguous k order, same for A and B)
// A (16x32 MxK) row-major:  lane L: m=L&15, half=L>>4
//   elems 0..15 = K(k0 + 16*half + 0..15)
// B (32x16 KxN) from col-major [N][K]: lane L: n=L&15, half=L>>4
//   elems 0..15 = K(k0 + 16*half + 0..15) of column n
// ---------------------------------------------------------------------------
__device__ __forceinline__ v16h frag_a(const _Float16* __restrict__ rowp, int half) {
    v16h a;
    a = *(const v16h*)(rowp + half * 16);
    return a;
}
__device__ __forceinline__ v16h frag_b(const _Float16* __restrict__ colp, int half) {
    return *(const v16h*)(colp + half * 16);
}
#define WMMA_F16(a, b, c) \
    __builtin_amdgcn_wmma_f32_16x16x32_f16(false, (a), false, (b), (short)0, (c), false, false)

// ---------------------------------------------------------------------------
// One-time: convert f32 [K,N] row-major weight -> f16 [N,K] (column-major)
// ---------------------------------------------------------------------------
__global__ void wt_transpose_f16(const float* __restrict__ src,
                                 _Float16* __restrict__ dst, int K, int N) {
    int idx = blockIdx.x * blockDim.x + threadIdx.x;
    if (idx >= K * N) return;
    int k = idx % K;
    int n = idx / K;
    dst[(size_t)n * K + k] = (_Float16)src[(size_t)k * N + n];
}

// ---------------------------------------------------------------------------
// Per-launch init: x16 = f16(inputs); s0 = 0; out[:,0,:] = inputs; *bar = 0
// ---------------------------------------------------------------------------
__global__ void rnn_init(const float* __restrict__ inputs,
                         _Float16* __restrict__ x16,
                         _Float16* __restrict__ s0,
                         float* __restrict__ out,
                         unsigned* __restrict__ bar,
                         int B, int D, int U, int T) {
    int idx = blockIdx.x * blockDim.x + threadIdx.x;
    if (idx == 0) *bar = 0u;
    if (idx < B * U) s0[idx] = (_Float16)0.0f;
    if (idx < B * D) {
        float v = inputs[idx];
        x16[idx] = (_Float16)v;
        int b = idx / D, d = idx % D;
        out[(size_t)b * T * D + d] = v;   // timestep 0 = raw input
    }
}

// ---------------------------------------------------------------------------
// Device-wide software barrier (all blocks co-resident; ticket/epoch scheme)
// ---------------------------------------------------------------------------
__device__ __forceinline__ void grid_barrier(unsigned* bar, unsigned target) {
    __threadfence();                       // release my writes at agent scope
    __syncthreads();
    if (threadIdx.x == 0) {
        __hip_atomic_fetch_add(bar, 1u, __ATOMIC_ACQ_REL, __HIP_MEMORY_SCOPE_AGENT);
        while (__hip_atomic_load(bar, __ATOMIC_ACQUIRE, __HIP_MEMORY_SCOPE_AGENT) < target)
            __builtin_amdgcn_s_sleep(1);
    }
    __syncthreads();
    __threadfence();                       // acquire side
}

// ---------------------------------------------------------------------------
// Persistent fused RNN for the canonical shape B=128, D=512, U=1024.
// 128 blocks x 128 threads = 512 waves, 1 wave/SIMD (full VGPR budget).
// Each wave owns one 16x16 phase-A tile (and, for gw<256, one phase-B tile)
// for ALL timesteps; weight B-fragments live in registers (80 x v16h = 640
// VGPRs, loaded from L2 once). Per step the dependent work is just: load f16
// state fragments -> WMMAs -> tanh -> store, with 2 device barriers.
// With 1 wave/SIMD there is no co-resident wave to hide XDL latency, so the
// K accumulation is split into independent chains (3 in phase A, 2 in phase
// B) that the scheduler can interleave, then reduced with v_add_f32.
// ---------------------------------------------------------------------------
#define CB 128
#define CD 512
#define CU 1024

__global__ void __launch_bounds__(128, 1)
rnn_persistent(const _Float16* __restrict__ WiT,  // [U][D]
               const _Float16* __restrict__ WrT,  // [U][U]
               const _Float16* __restrict__ WoT,  // [D][U]
               const float*    __restrict__ br,   // [U]
               const float*    __restrict__ bo,   // [D]
               _Float16*       __restrict__ xbuf, // [B,D] f16 state (y feeds back)
               _Float16*       __restrict__ s0,   // [B,U] f16 state ping
               _Float16*       __restrict__ s1,   // [B,U] f16 state pong
               float*          __restrict__ out,  // [B,T,D]
               int T, unsigned* __restrict__ bar) {
    const int lane = threadIdx.x & 31;
    const int gw   = blockIdx.x * 4 + (threadIdx.x >> 5);   // 0..511
    const int half = lane >> 4;
    const int l15  = lane & 15;

    // Phase A tile (m-major over 8 x 64 tiles)
    const int am0 = (gw & 7) << 4;           // batch-tile origin
    const int an0 = (gw >> 3) << 4;          // U-column origin
    // Phase B tile (only waves gw<256 participate; 8 x 32 tiles)
    const int bm0 = am0;
    const int bn0 = ((gw < 256 ? gw : 0) >> 3) << 4;  // clamped: loads stay in-range

    // ---- Preload all weight B-fragments into registers (once) ----
    v16h wA[48];
    {
        const _Float16* ci = WiT + (size_t)(an0 + l15) * CD;
#pragma unroll
        for (int j = 0; j < 16; ++j) wA[j] = frag_b(ci + j * 32, half);
        const _Float16* cr = WrT + (size_t)(an0 + l15) * CU;
#pragma unroll
        for (int j = 0; j < 32; ++j) wA[16 + j] = frag_b(cr + j * 32, half);
    }
    v16h wB[32];
    {
        const _Float16* co = WoT + (size_t)(bn0 + l15) * CU;
#pragma unroll
        for (int j = 0; j < 32; ++j) wB[j] = frag_b(co + j * 32, half);
    }
    const float bvA = br[an0 + l15];
    const float bvB = bo[bn0 + l15];

    unsigned target = 0;
    const unsigned nb = gridDim.x;

    for (int t = 1; t < T; ++t) {
        const _Float16* scur = (t & 1) ? s0 : s1;
        _Float16*       snxt = (t & 1) ? s1 : s0;

        // ---- Phase A: h = tanh(x @ Wi + s @ Wr + br) ----
        // Three independent 16-chunk WMMA chains (x, s-low, s-high) for ILP.
        {
            v8f a0 = {}, a1 = {}, a2 = {};
            const _Float16* arow = xbuf + (size_t)(am0 + l15) * CD;
            const _Float16* srow = scur + (size_t)(am0 + l15) * CU;
#pragma unroll
            for (int j = 0; j < 16; ++j) {
                a0 = WMMA_F16(frag_a(arow + j * 32, half),          wA[j],      a0);
                a1 = WMMA_F16(frag_a(srow + j * 32, half),          wA[16 + j], a1);
                a2 = WMMA_F16(frag_a(srow + 512 + j * 32, half),    wA[32 + j], a2);
            }
            _Float16* hp = snxt + (size_t)(am0 + 8 * half) * CU + (an0 + l15);
#pragma unroll
            for (int i = 0; i < 8; ++i)
                hp[(size_t)i * CU] = (_Float16)tanhf(a0[i] + a1[i] + a2[i] + bvA);
        }

        target += nb;
        grid_barrier(bar, target);

        // ---- Phase B: y = tanh(h @ Wo + bo); wave-uniform predicate ----
        // Two independent 16-chunk WMMA chains.
        if (gw < 256) {
            v8f a0 = {}, a1 = {};
            const _Float16* arow = snxt + (size_t)(bm0 + l15) * CU;
#pragma unroll
            for (int j = 0; j < 16; ++j) {
                a0 = WMMA_F16(frag_a(arow + j * 32, half),       wB[j],      a0);
                a1 = WMMA_F16(frag_a(arow + 512 + j * 32, half), wB[16 + j], a1);
            }
            const int col = bn0 + l15;
#pragma unroll
            for (int i = 0; i < 8; ++i) {
                int   row = bm0 + 8 * half + i;          // batch index
                float v   = tanhf(a0[i] + a1[i] + bvB);
                xbuf[(size_t)row * CD + col]                     = (_Float16)v;
                out[(size_t)row * T * CD + (size_t)t * CD + col] = v;
            }
        }

        target += nb;
        grid_barrier(bar, target);
    }
}

// ---------------------------------------------------------------------------
// Generic fallback kernels (non-canonical shapes): one wave per 16x16 tile
// ---------------------------------------------------------------------------
__global__ void __launch_bounds__(256)
rnn_phase_a(const _Float16* __restrict__ x, const _Float16* __restrict__ s,
            const _Float16* __restrict__ WiT, const _Float16* __restrict__ WrT,
            const float* __restrict__ bias, _Float16* __restrict__ h,
            int B, int D, int U) {
    const int lane = threadIdx.x & 31;
    const int wid  = (blockIdx.x * blockDim.x + threadIdx.x) >> 5;
    const int mT = B >> 4, nT = U >> 4;
    if (wid >= mT * nT) return;
    const int m0 = (wid % mT) << 4, n0 = (wid / mT) << 4;
    const int half = lane >> 4, l15 = lane & 15;
    v8f acc = {};
    {
        const _Float16* arow = x + (size_t)(m0 + l15) * D;
        const _Float16* bcol = WiT + (size_t)(n0 + l15) * D;
        for (int k0 = 0; k0 < D; k0 += 32)
            acc = WMMA_F16(frag_a(arow + k0, half), frag_b(bcol + k0, half), acc);
    }
    {
        const _Float16* arow = s + (size_t)(m0 + l15) * U;
        const _Float16* bcol = WrT + (size_t)(n0 + l15) * U;
        for (int k0 = 0; k0 < U; k0 += 32)
            acc = WMMA_F16(frag_a(arow + k0, half), frag_b(bcol + k0, half), acc);
    }
    const float bv = bias[n0 + l15];
    _Float16* hp = h + (size_t)(m0 + 8 * half) * U + (n0 + l15);
#pragma unroll
    for (int i = 0; i < 8; ++i) hp[(size_t)i * U] = (_Float16)tanhf(acc[i] + bv);
}

__global__ void __launch_bounds__(256)
rnn_phase_b(const _Float16* __restrict__ h, const _Float16* __restrict__ WoT,
            const float* __restrict__ bias, _Float16* __restrict__ y16,
            float* __restrict__ out, int B, int D, int U, int T, int t) {
    const int lane = threadIdx.x & 31;
    const int wid  = (blockIdx.x * blockDim.x + threadIdx.x) >> 5;
    const int mT = B >> 4, nT = D >> 4;
    if (wid >= mT * nT) return;
    const int m0 = (wid % mT) << 4, n0 = (wid / mT) << 4;
    const int half = lane >> 4, l15 = lane & 15;
    v8f acc = {};
    const _Float16* arow = h + (size_t)(m0 + l15) * U;
    const _Float16* bcol = WoT + (size_t)(n0 + l15) * U;
    for (int k0 = 0; k0 < U; k0 += 32)
        acc = WMMA_F16(frag_a(arow + k0, half), frag_b(bcol + k0, half), acc);
    const float bv = bias[n0 + l15];
    const int col = n0 + l15;
#pragma unroll
    for (int i = 0; i < 8; ++i) {
        int row = m0 + 8 * half + i;
        float v = tanhf(acc[i] + bv);
        y16[(size_t)row * D + col] = (_Float16)v;
        out[(size_t)row * T * D + (size_t)t * D + col] = v;
    }
}

// ---------------------------------------------------------------------------
extern "C" void kernel_launch(void* const* d_in, const int* in_sizes, int n_in,
                              void* d_out, int out_size, void* d_ws, size_t ws_size,
                              hipStream_t stream) {
    const float* inputs = (const float*)d_in[0];
    const float* Wi     = (const float*)d_in[1];   // [D,U]
    const float* Wr     = (const float*)d_in[2];   // [U,U]
    const float* br     = (const float*)d_in[3];   // [U]
    const float* Wo     = (const float*)d_in[4];   // [U,D]
    const float* bo     = (const float*)d_in[5];   // [D]
    float*       out    = (float*)d_out;

    const int U = in_sizes[3];            // 1024
    const int D = in_sizes[5];            // 512
    const int B = in_sizes[0] / D;        // 128
    const int T = out_size / (B * D);     // 512

    // Workspace carving (256B aligned blocks)
    char*  ws  = (char*)d_ws;
    size_t off = 0;
    auto carve = [&](size_t bytes) -> void* {
        void* p = ws + off;
        off += (bytes + 255) & ~(size_t)255;
        return p;
    };
    _Float16* WiT   = (_Float16*)carve((size_t)U * D * sizeof(_Float16));
    _Float16* WrT   = (_Float16*)carve((size_t)U * U * sizeof(_Float16));
    _Float16* WoT   = (_Float16*)carve((size_t)D * U * sizeof(_Float16));
    _Float16* xbuf  = (_Float16*)carve((size_t)B * D * sizeof(_Float16));
    _Float16* sbuf0 = (_Float16*)carve((size_t)B * U * sizeof(_Float16));
    _Float16* sbuf1 = (_Float16*)carve((size_t)B * U * sizeof(_Float16));
    unsigned* bar   = (unsigned*)carve(256);
    (void)ws_size;

    // One-time weight convert+transpose (f32 row-major -> f16 col-major)
    {
        int n;
        n = D * U; wt_transpose_f16<<<(n + 255) / 256, 256, 0, stream>>>(Wi, WiT, D, U);
        n = U * U; wt_transpose_f16<<<(n + 255) / 256, 256, 0, stream>>>(Wr, WrT, U, U);
        n = U * D; wt_transpose_f16<<<(n + 255) / 256, 256, 0, stream>>>(Wo, WoT, U, D);
    }
    // State init + timestep-0 output + barrier reset
    {
        int n = B * U;   // >= B*D
        rnn_init<<<(n + 255) / 256, 256, 0, stream>>>(inputs, xbuf, sbuf0, out, bar,
                                                      B, D, U, T);
    }

    if (B == CB && D == CD && U == CU) {
        // Persistent fused path: one kernel for all T-1 steps.
        rnn_persistent<<<128, 128, 0, stream>>>(WiT, WrT, WoT, br, bo,
                                                xbuf, sbuf0, sbuf1, out, T, bar);
    } else {
        // Fallback: per-step kernel pair.
        const int wavesA  = (B >> 4) * (U >> 4);
        const int wavesB  = (B >> 4) * (D >> 4);
        const int blocksA = (wavesA + 7) / 8;
        const int blocksB = (wavesB + 7) / 8;
        _Float16* scur = sbuf0;
        _Float16* snxt = sbuf1;
        for (int t = 1; t < T; ++t) {
            rnn_phase_a<<<blocksA, 256, 0, stream>>>(xbuf, scur, WiT, WrT, br, snxt, B, D, U);
            rnn_phase_b<<<blocksB, 256, 0, stream>>>(snxt, WoT, bo, xbuf, out, B, D, U, T, t);
            _Float16* tmp = scur; scur = snxt; snxt = tmp;
        }
    }
}
